// GCNeXtBlock_2061584302457
// MI455X (gfx1250) — matrix operand order repaired and checked by reading, not compiled, not run on hardware
//
#include <hip/hip_runtime.h>

// ---------------- types ----------------
typedef __attribute__((ext_vector_type(16))) __bf16      v16bf;
typedef __attribute__((ext_vector_type(8)))  float       v8f;
typedef __attribute__((ext_vector_type(4)))  unsigned    u32x4;
typedef int v4i_ __attribute__((vector_size(16)));       // matches async-to-LDS builtin pointee

union Frag { u32x4 u[2]; v16bf v; };

// ---------------- helpers ----------------
__device__ __forceinline__ unsigned short f2bf(float f) {
    unsigned u = __float_as_uint(f);
    u += 0x7FFFu + ((u >> 16) & 1u);           // round-to-nearest-even
    return (unsigned short)(u >> 16);
}
__device__ __forceinline__ float bf2f(unsigned short h) {
    return __uint_as_float(((unsigned)h) << 16);
}
// order-preserving float -> uint key for atomicMax over signed floats
__device__ __forceinline__ unsigned fkey(float f) {
    unsigned u = __float_as_uint(f);
    return (u & 0x80000000u) ? ~u : (u | 0x80000000u);
}
__device__ __forceinline__ float fdecode(unsigned k) {
    unsigned u = (k & 0x80000000u) ? (k ^ 0x80000000u) : ~k;
    return __uint_as_float(u);
}
// A fragment: two 16B chunks, 16 elements apart (K-halves per lane)
__device__ __forceinline__ void loadA(Frag& f, const unsigned short* p) {
    f.u[0] = *(const u32x4*)(p);
    f.u[1] = *(const u32x4*)(p + 16);
}
// B fragment: two 16B chunks, 16 elements apart (K-halves per lane)
__device__ __forceinline__ void loadB(Frag& f, const unsigned short* p) {
    f.u[0] = *(const u32x4*)(p);
    f.u[1] = *(const u32x4*)(p + 16);
}

// stage `nvec` 16B chunks from global to LDS via the async engine (gfx1250 path)
__device__ __forceinline__ void async_stage(const void* gsrc, void* ldst, int nvec,
                                            int tid, int nthreads) {
#if __has_builtin(__builtin_amdgcn_global_load_async_to_lds_b128)
    auto* g = (__attribute__((address_space(1))) v4i_*)gsrc;
    auto* l = (__attribute__((address_space(3))) v4i_*)ldst;
    for (int i = tid; i < nvec; i += nthreads)
        __builtin_amdgcn_global_load_async_to_lds_b128(g + i, l + i, 0, 0);
#if __has_builtin(__builtin_amdgcn_s_wait_asynccnt)
    __builtin_amdgcn_s_wait_asynccnt(0);
#else
    asm volatile("s_wait_asynccnt 0x0" ::: "memory");
#endif
#else
    const float* s = (const float*)gsrc;
    float* d = (float*)ldst;
    for (int i = tid; i < nvec * 4; i += nthreads) d[i] = s[i];
#endif
}

// ---------------- elementwise: f32 -> bf16 (weights) ----------------
__global__ void k_cvt(const float* __restrict__ in, unsigned short* __restrict__ out, int n) {
    int i = blockIdx.x * blockDim.x + threadIdx.x;
    if (i < n) out[i] = f2bf(in[i]);
}

// ---------------- x [B,256,2048] f32 -> xT [B*T,256] bf16 + sq[b,t] ----------------
__global__ void k_xt_sq(const float* __restrict__ x, unsigned short* __restrict__ xT,
                        float* __restrict__ sq) {
    int bt = blockIdx.x * blockDim.x + threadIdx.x;   // 32768 threads
    int b = bt >> 11, t = bt & 2047;
    const float* xp = x + (size_t)b * 256 * 2048 + t;
    unsigned short* op = xT + (size_t)bt * 256;
    float s = 0.f;
    for (int c = 0; c < 256; ++c) {
        float v = xp[(size_t)c * 2048];
        s += v * v;
        op[c] = f2bf(v);
    }
    sq[bt] = s;
}

// ---------------- fill u32 ----------------
__global__ void k_fill(unsigned* __restrict__ p, unsigned val, int n) {
    int i = blockIdx.x * blockDim.x + threadIdx.x;
    if (i < n) p[i] = val;
}

// ---------------- generic WMMA GEMM:  D[M,N] = A[M,K](bf16) * B[N,K]^T(bf16) ----------------
// 32(M) x 64(N) tile per wave, double-buffered K pipeline (K multiple of 64).
// EPI 0: relu -> bf16 ; EPI 1: (+bias) f32 ; EPI 2: (+bias) keyed atomicMax (max-over-K)
template <int EPI>
__global__ void __launch_bounds__(128, 1)
k_gemm(const unsigned short* __restrict__ A,
       const unsigned short* __restrict__ Bw,
       const float* __restrict__ bias,
       void* __restrict__ out,
       int M, int N, int K, int lda, int ldb, int boff) {
    int lane = threadIdx.x & 31;
    int gw = blockIdx.x * (blockDim.x >> 5) + (threadIdx.x >> 5);
    int nTiles = N >> 6;
    int totalW = (M >> 5) * nTiles;
    if (gw >= totalW) return;
    int mt = gw / nTiles, nt = gw % nTiles;
    int l15 = lane & 15, lh = lane >> 4;

    const unsigned short* ap0 = A + (size_t)(mt * 32 + l15) * lda + lh * 8;
    const unsigned short* ap1 = ap0 + (size_t)16 * lda;
    const unsigned short* bp[4];
#pragma unroll
    for (int j = 0; j < 4; ++j)
        bp[j] = Bw + (size_t)(nt * 64 + j * 16 + l15) * ldb + boff + lh * 8;

    v8f acc[8] = {};
    Frag a0[2], a1[2], b0[4], b1[4];

    loadA(a0[0], ap0); loadA(a0[1], ap1);
#pragma unroll
    for (int j = 0; j < 4; ++j) loadB(b0[j], bp[j]);

    for (int k0 = 0; k0 < K; k0 += 64) {
        // stage B: issue loads for k0+32 while computing on buffer 0
        loadA(a1[0], ap0 + k0 + 32); loadA(a1[1], ap1 + k0 + 32);
#pragma unroll
        for (int j = 0; j < 4; ++j) loadB(b1[j], bp[j] + k0 + 32);
        __builtin_prefetch(ap0 + k0 + 128, 0, 0);
        __builtin_prefetch(ap1 + k0 + 128, 0, 0);
#pragma unroll
        for (int ms = 0; ms < 2; ++ms)
#pragma unroll
            for (int j = 0; j < 4; ++j)
                acc[ms * 4 + j] = __builtin_amdgcn_wmma_f32_16x16x32_bf16(
                    false, a0[ms].v, false, b0[j].v, (short)0, acc[ms * 4 + j], false, false);

        // stage A': issue loads for k0+64 (tail over-read lands in adjacent ws buffer, unused)
        loadA(a0[0], ap0 + k0 + 64); loadA(a0[1], ap1 + k0 + 64);
#pragma unroll
        for (int j = 0; j < 4; ++j) loadB(b0[j], bp[j] + k0 + 64);
#pragma unroll
        for (int ms = 0; ms < 2; ++ms)
#pragma unroll
            for (int j = 0; j < 4; ++j)
                acc[ms * 4 + j] = __builtin_amdgcn_wmma_f32_16x16x32_bf16(
                    false, a1[ms].v, false, b1[j].v, (short)0, acc[ms * 4 + j], false, false);
    }

#pragma unroll
    for (int j = 0; j < 4; ++j) {
        int col = nt * 64 + j * 16 + l15;
        float bv = bias ? bias[col] : 0.f;
#pragma unroll
        for (int ms = 0; ms < 2; ++ms) {
#pragma unroll
            for (int r = 0; r < 8; ++r) {
                int row = mt * 32 + ms * 16 + r + (lh << 3);
                float v = acc[ms * 4 + j][r] + bv;
                if (EPI == 0) {
                    v = v > 0.f ? v : 0.f;
                    ((unsigned short*)out)[(size_t)row * N + col] = f2bf(v);
                } else if (EPI == 1) {
                    ((float*)out)[(size_t)row * N + col] = v;
                } else {
                    int b = row / 6144;        // rows are p = b*6144 + t*3 + k
                    int r2 = row - b * 6144;
                    int t = r2 / 3;
                    atomicMax((unsigned*)out + (((size_t)b * 256 + col) * 2048 + t), fkey(v));
                }
            }
        }
    }
}

// ---------------- KNN: per wave = 16 rows; Gram via WMMA, double-buffered col tiles ----------------
__global__ void __launch_bounds__(32, 1)
k_knn(const unsigned short* __restrict__ xT, const float* __restrict__ sq,
      int* __restrict__ knn) {
    __shared__ float sc[16 * 17];
    int lane = threadIdx.x;                    // blockDim = 32
    int b = blockIdx.x >> 7;                   // T/16 = 128 row tiles
    int mt = blockIdx.x & 127;
    int l15 = lane & 15, lh = lane >> 4;

    const unsigned short* arow = xT + (size_t)(b * 2048 + mt * 16 + l15) * 256 + lh * 8;
    Frag af[8];
#pragma unroll
    for (int kk = 0; kk < 8; ++kk) loadA(af[kk], arow + kk * 32);

    const unsigned short* bbase = xT + (size_t)(b * 2048 + l15) * 256 + lh * 8;
    Frag bb[2][8];

    float v0 = -3e38f, v1 = -3e38f, v2 = -3e38f;
    int i0 = 0, i1 = 0, i2 = 0;

    auto loadBT = [&](Frag* dst, int nt) {
        const unsigned short* p = bbase + (size_t)nt * 16 * 256;
#pragma unroll
        for (int kk = 0; kk < 8; ++kk) loadB(dst[kk], p + kk * 32);
    };
    auto process = [&](const Frag* bf, int nt) {
        v8f acc = {};
#pragma unroll
        for (int kk = 0; kk < 8; ++kk)
            acc = __builtin_amdgcn_wmma_f32_16x16x32_bf16(
                false, af[kk].v, false, bf[kk].v, (short)0, acc, false, false);
        float sqv = sq[b * 2048 + nt * 16 + l15];
#pragma unroll
        for (int r = 0; r < 8; ++r)
            sc[(r + (lh << 3)) * 17 + l15] = 2.f * acc[r] - sqv;
        __syncthreads();
        if (lane < 16) {
            for (int j = 0; j < 16; ++j) {
                float v = sc[lane * 17 + j];
                int id = nt * 16 + j;
                if (v > v0)      { v2 = v1; i2 = i1; v1 = v0; i1 = i0; v0 = v; i0 = id; }
                else if (v > v1) { v2 = v1; i2 = i1; v1 = v;  i1 = id; }
                else if (v > v2) { v2 = v;  i2 = id; }
            }
        }
        __syncthreads();
    };

    loadBT(bb[0], 0);
    for (int nt = 0; nt < 128; nt += 2) {
        loadBT(bb[1], nt + 1);                 // in flight during process of bb[0]
        process(bb[0], nt);
        loadBT(bb[0], nt + 2);                 // nt+2==128 over-read stays inside d_ws, unused
        process(bb[1], nt + 1);
    }

    if (lane < 16) {
        int t = b * 2048 + mt * 16 + lane;
        knn[t * 3 + 0] = i0; knn[t * 3 + 1] = i1; knn[t * 3 + 2] = i2;
    }
}

// ---------------- grouped temporal conv k=3, pad=1, groups=32 (128ch) ----------------
__global__ void k_tconv(const unsigned short* __restrict__ y1T, const float* __restrict__ w,
                        const float* __restrict__ bias, unsigned short* __restrict__ y2T) {
    __shared__ float wsm[1536];                // tw2 [128,4,3]
    int o = threadIdx.x;                       // 128
    async_stage(w, wsm, 384, o, 128);          // 6144B via async engine
    __syncthreads();

    int base = blockIdx.x * 8;                 // grid = 4096
    float wr[12];
#pragma unroll
    for (int i = 0; i < 12; ++i) wr[i] = wsm[o * 12 + i];
    float bv = bias[o];
    int g4 = (o >> 2) << 2;
    for (int q = 0; q < 8; ++q) {
        int bt = base + q;
        int b = bt >> 11, t = bt & 2047;
        float acc = bv;
#pragma unroll
        for (int dt = 0; dt < 3; ++dt) {
            int tt = t + dt - 1;
            if (tt < 0 || tt >= 2048) continue;
            size_t row = (size_t)((b << 11) + tt) * 128;
#pragma unroll
            for (int ci = 0; ci < 4; ++ci)
                acc += wr[ci * 3 + dt] * bf2f(y1T[row + g4 + ci]);
        }
        acc = acc > 0.f ? acc : 0.f;
        y2T[(size_t)bt * 128 + o] = f2bf(acc);
    }
}

// ---------------- graph feature: s1 = relu(U[idx] + V) ----------------
__global__ void k_gather(const float* __restrict__ UT, const float* __restrict__ VT,
                         const int* __restrict__ knn, unsigned short* __restrict__ s1T) {
    int c = threadIdx.x;                       // 128
    int p = blockIdx.x;                        // 98304 = B*T*3
    int bt = p / 3;
    int b = bt >> 11;
    int nb = knn[p];                           // neighbor t within batch
    float u = UT[((size_t)(b << 11) + nb) * 128 + c];
    float v = VT[(size_t)bt * 128 + c];
    float r = u + v;
    r = r > 0.f ? r : 0.f;
    s1T[(size_t)p * 128 + c] = f2bf(r);
}

// ---------------- grouped semantic conv (1,3) over K dim, pad 1 ----------------
__global__ void k_sconv(const unsigned short* __restrict__ s1T, const float* __restrict__ w,
                        const float* __restrict__ bias, unsigned short* __restrict__ s2T) {
    __shared__ float wsm[1536];                // sw2 [128,4,1,3]
    int o = threadIdx.x;                       // 128
    async_stage(w, wsm, 384, o, 128);
    __syncthreads();

    int base = blockIdx.x * 8;                 // grid = 12288
    float wr[12];
#pragma unroll
    for (int i = 0; i < 12; ++i) wr[i] = wsm[o * 12 + i];
    float bv = bias[o];
    int g4 = (o >> 2) << 2;
    for (int q = 0; q < 8; ++q) {
        int p = base + q;
        int k = p % 3;
        int pb = p - k;                        // base row of this t's triple
        float acc = bv;
#pragma unroll
        for (int dk = 0; dk < 3; ++dk) {
            int kk = k + dk - 1;
            if (kk < 0 || kk >= 3) continue;
            size_t row = (size_t)(pb + kk) * 128;
#pragma unroll
            for (int ci = 0; ci < 4; ++ci)
                acc += wr[ci * 3 + dk] * bf2f(s1T[row + g4 + ci]);
        }
        acc = acc > 0.f ? acc : 0.f;
        s2T[(size_t)p * 128 + o] = f2bf(acc);
    }
}

// ---------------- final: out = relu(tout + 2x + sout) ----------------
__global__ void k_combine(const float* __restrict__ toutT, const float* __restrict__ x,
                          const unsigned* __restrict__ smax, float* __restrict__ out) {
    size_t i = (size_t)blockIdx.x * blockDim.x + threadIdx.x;   // 8388608, [b,o,t] order
    int t = (int)(i & 2047);
    int bo = (int)(i >> 11);
    int o = bo & 255, b = bo >> 8;
    float sm = fdecode(smax[i]);
    float tv = toutT[((size_t)(b << 11) + t) * 256 + o];
    float r = tv + 2.f * x[i] + sm;
    out[i] = r > 0.f ? r : 0.f;
}

// ---------------- launcher ----------------
extern "C" void kernel_launch(void* const* d_in, const int* in_sizes, int n_in,
                              void* d_out, int out_size, void* d_ws, size_t ws_size,
                              hipStream_t stream) {
    const float* x   = (const float*)d_in[0];
    const float* tw1 = (const float*)d_in[1];
    const float* tb1 = (const float*)d_in[2];
    const float* tw2 = (const float*)d_in[3];
    const float* tb2 = (const float*)d_in[4];
    const float* tw3 = (const float*)d_in[5];
    const float* tb3 = (const float*)d_in[6];
    const float* sw1 = (const float*)d_in[7];
    const float* sb1 = (const float*)d_in[8];
    const float* sw2 = (const float*)d_in[9];
    const float* sb2 = (const float*)d_in[10];
    const float* sw3 = (const float*)d_in[11];
    const float* sb3 = (const float*)d_in[12];

    char* w = (char*)d_ws;
    size_t off = 0;
    auto alloc = [&](size_t bytes) -> void* {
        void* p = w + off;
        off = (off + bytes + 255) & ~(size_t)255;
        return p;
    };
    unsigned short* tw1b = (unsigned short*)alloc(32768u * 2);       // [128,256]
    unsigned short* tw3b = (unsigned short*)alloc(32768u * 2);       // [256,128]
    unsigned short* sw1b = (unsigned short*)alloc(65536u * 2);       // [128,512]
    unsigned short* sw3b = (unsigned short*)alloc(32768u * 2);       // [256,128]
    unsigned short* xT   = (unsigned short*)alloc((size_t)32768 * 256 * 2);
    float*          sq   = (float*)alloc(32768u * 4);
    int*            knn  = (int*)alloc((size_t)32768 * 3 * 4);
    unsigned short* y1T  = (unsigned short*)alloc((size_t)32768 * 128 * 2);
    unsigned short* y2T  = (unsigned short*)alloc((size_t)32768 * 128 * 2);
    float*          toutT= (float*)alloc((size_t)32768 * 256 * 4);
    float*          UT   = (float*)alloc((size_t)32768 * 128 * 4);
    float*          VT   = (float*)alloc((size_t)32768 * 128 * 4);
    unsigned short* s1T  = (unsigned short*)alloc((size_t)98304 * 128 * 2);
    unsigned short* s2T  = (unsigned short*)alloc((size_t)98304 * 128 * 2);
    unsigned*       smax = (unsigned*)alloc((size_t)8388608 * 4);

    // weight conversions
    k_cvt<<<128, 256, 0, stream>>>(tw1, tw1b, 32768);
    k_cvt<<<128, 256, 0, stream>>>(tw3, tw3b, 32768);
    k_cvt<<<256, 256, 0, stream>>>(sw1, sw1b, 65536);
    k_cvt<<<128, 256, 0, stream>>>(sw3, sw3b, 32768);

    // transpose + squared norms
    k_xt_sq<<<128, 256, 0, stream>>>(x, xT, sq);

    // temporal: 1x1 -> grouped k3 -> 1x1
    k_gemm<0><<<512, 128, 0, stream>>>(xT, tw1b, tb1, y1T, 32768, 128, 256, 256, 256, 0);
    k_tconv<<<4096, 128, 0, stream>>>(y1T, tw2, tb2, y2T);
    k_gemm<1><<<1024, 128, 0, stream>>>(y2T, tw3b, tb3, toutT, 32768, 256, 128, 128, 128, 0);

    // knn on Gram matrix
    k_knn<<<2048, 32, 0, stream>>>(xT, sq, knn);

    // semantic: split W1 into neighbor/center halves
    k_gemm<1><<<512, 128, 0, stream>>>(xT, sw1b, nullptr, UT, 32768, 128, 256, 256, 512, 0);
    k_gemm<1><<<512, 128, 0, stream>>>(xT, sw1b, sb1,     VT, 32768, 128, 256, 256, 512, 256);
    k_gather<<<98304, 128, 0, stream>>>(UT, VT, knn, s1T);
    k_sconv<<<12288, 128, 0, stream>>>(s1T, sw2, sb2, s2T);

    // s3 GEMM with fused max-over-K via keyed atomicMax
    k_fill<<<32768, 256, 0, stream>>>(smax, 0u, 8388608);
    k_gemm<2><<<3072, 128, 0, stream>>>(s2T, sw3b, sb3, smax, 98304, 256, 128, 128, 128, 0);

    // final combine
    k_combine<<<32768, 256, 0, stream>>>(toutT, x, smax, (float*)d_out);
}
